// CapNet_3315714752682
// MI455X (gfx1250) — matrix operand order repaired and checked by reading, not compiled, not run on hardware
//
#include <hip/hip_runtime.h>
#include <hip/hip_bf16.h>

typedef __bf16 bf16_t;
typedef __attribute__((ext_vector_type(16))) __bf16 v16bf;
typedef __attribute__((ext_vector_type(8)))  __bf16 v8bf;
typedef __attribute__((ext_vector_type(8)))  float  v8f;

#define L_DIM 16
#define P_DIM 1024
#define DV    64
#define DP    16
#define B_DIM 128
#define PQ    (P_DIM*DP)   /* 16384 */

// ---------------- prep: W (l,p,v,q) f32 -> bf16 in two GEMM-friendly layouts
// Wbf [l][v][p*16+q] : B-matrix for s-GEMM  (K=(p,q) contiguous per column v)
// Wbf2[l][p][q][v]   : B-matrix for beta-GEMM (K=v contiguous per column q)
__global__ void __launch_bounds__(256)
k_wconv(const float* __restrict__ W, bf16_t* __restrict__ Wbf, bf16_t* __restrict__ Wbf2) {
    int idx = blockIdx.x * 256 + threadIdx.x;        // < 16777216, == W flat index
    int q = idx & 15;
    int v = (idx >> 4) & 63;
    int p = (idx >> 10) & 1023;
    int l = idx >> 20;
    bf16_t h = (bf16_t)W[idx];
    Wbf [(((l << 6) + v) << 14) + (p << 4) + q] = h;
    Wbf2[(((((l << 10) + p) << 4) + q) << 6) + v] = h;
}

// ---------------- gather features -> xbf[b][p][q] (bf16); zero s accumulator
__global__ void __launch_bounds__(256)
k_gather(const float* __restrict__ feature, const int* __restrict__ sel,
         bf16_t* __restrict__ xbf, float* __restrict__ s) {
    int idx = blockIdx.x * 256 + threadIdx.x;        // < 2097152
    int q  = idx & 15;
    int bp = idx >> 4;                                // b*1024+p
    int b  = bp >> 10;
    int row = sel[bp] + ((b >= 64) ? 64 : 0);
    xbf[idx] = (bf16_t)feature[row * DP + q];
    if (idx < B_DIM * L_DIM * DV) s[idx] = 0.0f;
}

// ---------------- s[b,l,v] += sum_{p,q} (c[b,l,p]*x[b,p,q]) * W[l,p,v,q]
// grid = (kchunk=16, l=16), block = 256 (8 waves). wave w -> M-tile (b rows 16w..16w+15)
// Ping-pong software pipeline (unroll-2): compute set A while loading set B.
// UNIFORM is a compile-time flag (pass 1 uses c = 1/16 with no c loads).
template <int UNIFORM>
__global__ void __launch_bounds__(256)
k_sgemm(const bf16_t* __restrict__ xbf, const bf16_t* __restrict__ Wbf,
        const float* __restrict__ cbuf, float* __restrict__ s) {
    const int kc   = blockIdx.x;
    const int l    = blockIdx.y;
    const int wv   = threadIdx.x >> 5;
    const int lane = threadIdx.x & 31;
    const int hf   = lane >> 4;
    const int lr   = lane & 15;
    const int brow = (wv << 4) + lr;                 // A-matrix row (M)
    const int koff0 = hf ? 16 : 0;

    const bf16_t* xrow  = xbf + brow * PQ;
    const float*  crow  = cbuf + (((brow << 4) + l) << 10);
    const bf16_t* wbase = Wbf + ((l << 6) + lr) * PQ + (hf << 4);

    v8f acc[4] = {v8f{}, v8f{}, v8f{}, v8f{}};
    const int kbase0 = kc << 10;

    v8bf xA0, xA1, xB0{}, xB1{};
    float cA0, cA1, cB0 = 0.0f, cB1 = 0.0f;
    v16bf bA[4], bB[4] = {v16bf{}, v16bf{}, v16bf{}, v16bf{}};

    // ---- prologue: set A <- ks = 0
    xA0 = *(const v8bf*)(xrow + kbase0 + koff0);
    xA1 = *(const v8bf*)(xrow + kbase0 + koff0 + 8);
    cA0 = UNIFORM ? 0.0625f : crow[(kbase0 + koff0) >> 4];
    cA1 = UNIFORM ? 0.0625f : crow[(kbase0 + koff0 + 8) >> 4];
#pragma unroll
    for (int n = 0; n < 4; ++n)
        bA[n] = *(const v16bf*)(wbase + (n << 4) * PQ + kbase0);

    for (int ks = 0; ks < 32; ks += 2) {
        // ---- load set B <- ks+1 (ks <= 30, always valid)
        {
            const int kb1 = kbase0 + ((ks + 1) << 5);
            const int K = kb1 + koff0;
            xB0 = *(const v8bf*)(xrow + K);
            xB1 = *(const v8bf*)(xrow + K + 8);
            cB0 = UNIFORM ? 0.0625f : crow[K >> 4];
            cB1 = UNIFORM ? 0.0625f : crow[(K + 8) >> 4];
#pragma unroll
            for (int n = 0; n < 4; ++n)
                bB[n] = *(const v16bf*)(wbase + (n << 4) * PQ + kb1);
        }
        // ---- compute with set A (ks)
        {
            v16bf a;
#pragma unroll
            for (int i = 0; i < 8; ++i) {
                a[i]     = (bf16_t)((float)xA0[i] * cA0);
                a[i + 8] = (bf16_t)((float)xA1[i] * cA1);
            }
#pragma unroll
            for (int n = 0; n < 4; ++n)
                acc[n] = __builtin_amdgcn_wmma_f32_16x16x32_bf16(
                    false, a, false, bA[n], (short)0, acc[n], false, false);
        }
        // ---- load set A <- ks+2
        if (ks + 2 < 32) {
            const int kb2 = kbase0 + ((ks + 2) << 5);
            const int K = kb2 + koff0;
            xA0 = *(const v8bf*)(xrow + K);
            xA1 = *(const v8bf*)(xrow + K + 8);
            cA0 = UNIFORM ? 0.0625f : crow[K >> 4];
            cA1 = UNIFORM ? 0.0625f : crow[(K + 8) >> 4];
#pragma unroll
            for (int n = 0; n < 4; ++n)
                bA[n] = *(const v16bf*)(wbase + (n << 4) * PQ + kb2);
        }
        // ---- compute with set B (ks+1)
        {
            v16bf a;
#pragma unroll
            for (int i = 0; i < 8; ++i) {
                a[i]     = (bf16_t)((float)xB0[i] * cB0);
                a[i + 8] = (bf16_t)((float)xB1[i] * cB1);
            }
#pragma unroll
            for (int n = 0; n < 4; ++n)
                acc[n] = __builtin_amdgcn_wmma_f32_16x16x32_bf16(
                    false, a, false, bB[n], (short)0, acc[n], false, false);
        }
    }
    // ---- scatter partial sums:  C layout: lanes0-15 M=j, lanes16-31 M=j+8; N=lr
#pragma unroll
    for (int n = 0; n < 4; ++n)
#pragma unroll
        for (int j = 0; j < 8; ++j) {
            int bb = (wv << 4) + j + (hf << 3);
            int vv = (n << 4) + lr;
            atomicAdd(&s[((bb << 4) + l) * DV + vv], acc[n][j]);
        }
}

// ---------------- squash: v = squash(s); writes vbf (bf16) and zeroes s,
// or on the final pass writes caps + norms to d_out
__global__ void __launch_bounds__(64)
k_squash(float* __restrict__ s, bf16_t* __restrict__ vbf,
         float* __restrict__ caps_out, float* __restrict__ norms_out, int final_pass) {
    int bl = blockIdx.x;                              // b*16+l
    int v  = threadIdx.x;
    int idx = bl * DV + v;
    float sv = s[idx];
    float t = sv * sv;
#pragma unroll
    for (int off = 16; off > 0; off >>= 1) t += __shfl_down(t, off, 32);
    __shared__ float sh[2];
    if ((threadIdx.x & 31) == 0) sh[threadIdx.x >> 5] = t;
    __syncthreads();
    float sn = sh[0] + sh[1];
    float scale = (sn / (1.0f + sn)) * rsqrtf(fmaxf(sn, 1e-30f));
    float outv = sv * scale;
    if (final_pass) {
        caps_out[idx] = outv;
        if (v == 0) {
            int b = bl >> 4, l = bl & 15;
            norms_out[l * B_DIM + b] = sn / (1.0f + sn);   // ||squash(s)||
        }
    } else {
        vbf[idx] = (bf16_t)outv;
        s[idx] = 0.0f;                                // ready for next pass
    }
}

// ---------------- beta[b,l,p] (+)= sum_q x[b,p,q] * (sum_v v[b,l,v]*W[l,p,v,q])
// grid = (pchunk=16, l=16); block 256 (8 waves); wave w -> b rows 16w..16w+15
// Ping-pong software pipeline over p (unroll-2). FIRST is compile-time.
template <int FIRST>
__global__ void __launch_bounds__(256)
k_bupdate(const bf16_t* __restrict__ xbf, const bf16_t* __restrict__ Wbf2,
          const bf16_t* __restrict__ vbf, float* __restrict__ beta) {
    const int pc   = blockIdx.x;
    const int l    = blockIdx.y;
    const int wv   = threadIdx.x >> 5;
    const int lane = threadIdx.x & 31;
    const int hf   = lane >> 4;
    const int lr   = lane & 15;
    const int koff0 = hf ? 16 : 0;
    const int b0   = wv << 4;
    const int brow = b0 + lr;

    // A fragments: A[b, K=v] = v[b,l,v]; two K-steps (v 0..31, 32..63)
    v16bf a0, a1;
    {
        const bf16_t* vp = vbf + ((brow << 4) + l) * DV;
        v8bf g0 = *(const v8bf*)(vp + koff0);
        v8bf g1 = *(const v8bf*)(vp + koff0 + 8);
        v8bf g2 = *(const v8bf*)(vp + 32 + koff0);
        v8bf g3 = *(const v8bf*)(vp + 32 + koff0 + 8);
#pragma unroll
        for (int i = 0; i < 8; ++i) {
            a0[i] = g0[i]; a0[i + 8] = g1[i];
            a1[i] = g2[i]; a1[i + 8] = g3[i];
        }
    }

    const int pbase = pc << 6;
    const bf16_t* wcol = Wbf2 + ((((l << 10) << 4) + lr) << 6) + (hf << 4);

    // ---- prologue: set A <- pi = 0
    v16bf w0a = *(const v16bf*)(wcol + (pbase << 10));
    v16bf w1a = *(const v16bf*)(wcol + (pbase << 10) + 32);
    v16bf w0b{}, w1b{};
    float xva[8], xvb[8] = {0, 0, 0, 0, 0, 0, 0, 0};
#pragma unroll
    for (int j = 0; j < 8; ++j)
        xva[j] = (float)xbf[((b0 + j + (hf << 3)) * P_DIM + pbase) * DP + lr];

    for (int pi = 0; pi < 64; pi += 2) {
        const int p = pbase + pi;
        // ---- load set B <- p+1 (pi <= 62, always valid)
        {
            w0b = *(const v16bf*)(wcol + ((p + 1) << 10));
            w1b = *(const v16bf*)(wcol + ((p + 1) << 10) + 32);
#pragma unroll
            for (int j = 0; j < 8; ++j)
                xvb[j] = (float)xbf[((b0 + j + (hf << 3)) * P_DIM + p + 1) * DP + lr];
        }
        // ---- compute set A (p)
        {
            v8f g = v8f{};
            g = __builtin_amdgcn_wmma_f32_16x16x32_bf16(false, a0, false, w0a, (short)0, g, false, false);
            g = __builtin_amdgcn_wmma_f32_16x16x32_bf16(false, a1, false, w1a, (short)0, g, false, false);
#pragma unroll
            for (int j = 0; j < 8; ++j) {
                int bb = b0 + j + (hf << 3);
                float t = xva[j] * g[j];
#pragma unroll
                for (int off = 1; off < 16; off <<= 1) t += __shfl_xor(t, off, 32);
                if (lr == 0) {
                    float* dst = &beta[(((bb << 4) + l) << 10) + p];
                    if (FIRST) *dst = t; else *dst = *dst + t;
                }
            }
        }
        // ---- load set A <- p+2
        if (pi + 2 < 64) {
            w0a = *(const v16bf*)(wcol + ((p + 2) << 10));
            w1a = *(const v16bf*)(wcol + ((p + 2) << 10) + 32);
#pragma unroll
            for (int j = 0; j < 8; ++j)
                xva[j] = (float)xbf[((b0 + j + (hf << 3)) * P_DIM + p + 2) * DP + lr];
        }
        // ---- compute set B (p+1)
        {
            v8f g = v8f{};
            g = __builtin_amdgcn_wmma_f32_16x16x32_bf16(false, a0, false, w0b, (short)0, g, false, false);
            g = __builtin_amdgcn_wmma_f32_16x16x32_bf16(false, a1, false, w1b, (short)0, g, false, false);
#pragma unroll
            for (int j = 0; j < 8; ++j) {
                int bb = b0 + j + (hf << 3);
                float t = xvb[j] * g[j];
#pragma unroll
                for (int off = 1; off < 16; off <<= 1) t += __shfl_xor(t, off, 32);
                if (lr == 0) {
                    float* dst = &beta[(((bb << 4) + l) << 10) + p + 1];
                    if (FIRST) *dst = t; else *dst = *dst + t;
                }
            }
        }
    }
}

// ---------------- softmax over L for each (b,p)
__global__ void __launch_bounds__(256)
k_softmax(const float* __restrict__ beta, float* __restrict__ cbuf) {
    int idx = blockIdx.x * 256 + threadIdx.x;        // b*1024+p, < 131072
    int b = idx >> 10, p = idx & 1023;
    float vals[16];
    float mx = -1e30f;
#pragma unroll
    for (int l = 0; l < 16; ++l) {
        vals[l] = beta[(((b << 4) + l) << 10) + p];
        mx = fmaxf(mx, vals[l]);
    }
    float sum = 0.0f;
#pragma unroll
    for (int l = 0; l < 16; ++l) { vals[l] = __expf(vals[l] - mx); sum += vals[l]; }
    float inv = 1.0f / sum;
#pragma unroll
    for (int l = 0; l < 16; ++l) cbuf[(((b << 4) + l) << 10) + p] = vals[l] * inv;
}

extern "C" void kernel_launch(void* const* d_in, const int* in_sizes, int n_in,
                              void* d_out, int out_size, void* d_ws, size_t ws_size,
                              hipStream_t stream) {
    const float* feature = (const float*)d_in[0];
    const int*   sel     = (const int*)d_in[1];
    const float* W       = (const float*)d_in[2];
    float* out = (float*)d_out;

    char* ws = (char*)d_ws;
    bf16_t* Wbf  = (bf16_t*)(ws);                     // 33,554,432 B
    bf16_t* Wbf2 = (bf16_t*)(ws + 33554432);          // 33,554,432 B
    bf16_t* xbf  = (bf16_t*)(ws + 67108864);          //  4,194,304 B
    bf16_t* vbf  = (bf16_t*)(ws + 71303168);          //    262,144 B
    float*  s    = (float*)(ws + 71565312);           //    524,288 B
    float*  beta = (float*)(ws + 72089600);           //  8,388,608 B
    float*  cbuf = (float*)(ws + 80478208);           //  8,388,608 B  (total ~84.8 MB)

    dim3 g16(16, 16);

    k_wconv  <<<65536, 256, 0, stream>>>(W, Wbf, Wbf2);
    k_gather <<<8192,  256, 0, stream>>>(feature, sel, xbf, s);

    // iteration 1 (c uniform = 1/16)
    k_sgemm<1>  <<<g16,  256, 0, stream>>>(xbf, Wbf, cbuf, s);
    k_squash    <<<2048,  64, 0, stream>>>(s, vbf, nullptr, nullptr, 0);
    k_bupdate<1><<<g16,  256, 0, stream>>>(xbf, Wbf2, vbf, beta);
    k_softmax   <<<512,  256, 0, stream>>>(beta, cbuf);

    // iteration 2
    k_sgemm<0>  <<<g16,  256, 0, stream>>>(xbf, Wbf, cbuf, s);
    k_squash    <<<2048,  64, 0, stream>>>(s, vbf, nullptr, nullptr, 0);
    k_bupdate<0><<<g16,  256, 0, stream>>>(xbf, Wbf2, vbf, beta);
    k_softmax   <<<512,  256, 0, stream>>>(beta, cbuf);

    // final
    k_sgemm<0>  <<<g16,  256, 0, stream>>>(xbf, Wbf, cbuf, s);
    k_squash    <<<2048,  64, 0, stream>>>(s, vbf, out, out + B_DIM * L_DIM * DV, 1);
}
